// DecoderLayer_82386062671981
// MI455X (gfx1250) — compile-verified, matrix operand order checked by reading, not run on hardware
//
#include <hip/hip_runtime.h>
#include <hip/hip_bf16.h>
#include <math.h>

#define HS 4096
#define NH 32
#define HD 128
#define II 11008
#define SEQ 2048

typedef __attribute__((ext_vector_type(16))) __bf16 v16bf;
typedef __attribute__((ext_vector_type(8)))  __bf16 v8bf;
typedef __attribute__((ext_vector_type(8)))  float  v8f;

__device__ __forceinline__ v8f wmma_bf16(v16bf a, v16bf b, v8f c) {
  return __builtin_amdgcn_wmma_f32_16x16x32_bf16(false, a, false, b, (short)0, c,
                                                 false, false);
}

// ---------------------------------------------------------------------------
// fp32 -> bf16 cast (weights)
// ---------------------------------------------------------------------------
__global__ void cast_bf16_kernel(const float* __restrict__ in,
                                 __bf16* __restrict__ out, size_t n) {
  size_t i = (size_t)blockIdx.x * blockDim.x + threadIdx.x;
  size_t stride = (size_t)gridDim.x * blockDim.x;
  for (; i < n; i += stride) out[i] = (__bf16)in[i];
}

// ---------------------------------------------------------------------------
// LayerNorm (per row of `cols` fp32), bf16 output
// ---------------------------------------------------------------------------
__global__ __launch_bounds__(256) void ln_bf16_kernel(
    const float* __restrict__ x, const float* __restrict__ w,
    __bf16* __restrict__ out, int cols) {
  const int row = blockIdx.x;
  const int tid = threadIdx.x;
  const float* xr = x + (size_t)row * cols;
  __shared__ float red[256];

  float s = 0.f;
  for (int i = tid; i < cols; i += 256) s += xr[i];
  red[tid] = s;
  __syncthreads();
  for (int off = 128; off > 0; off >>= 1) {
    if (tid < off) red[tid] += red[tid + off];
    __syncthreads();
  }
  const float mean = red[0] / (float)cols;
  __syncthreads();

  float vs = 0.f;
  for (int i = tid; i < cols; i += 256) {
    float d = xr[i] - mean;
    vs += d * d;
  }
  red[tid] = vs;
  __syncthreads();
  for (int off = 128; off > 0; off >>= 1) {
    if (tid < off) red[tid] += red[tid + off];
    __syncthreads();
  }
  const float rstd = rsqrtf(red[0] / (float)cols + 1e-5f);

  __bf16* orow = out + (size_t)row * cols;
  for (int i = tid; i < cols; i += 256)
    orow[i] = (__bf16)((xr[i] - mean) * rstd * w[i]);
}

// ---------------------------------------------------------------------------
// Tiled WMMA GEMM: C[M,N] = A[M,K] * B[N,K]^T  (both bf16, fp32 accum)
// Block: 256 thr = 8 waves, 128x128 tile; wave tile 32x64 (2x4 of 16x16).
// Software-pipelined: next K-tile loaded to registers during WMMA block.
// MODE 0: bf16 out.  MODE 1: fp32 out.  MODE 2: fp32 out + fp32 residual R.
// ---------------------------------------------------------------------------
template <int MODE>
__global__ __launch_bounds__(256) void gemm_bf16_kernel(
    const __bf16* __restrict__ A, const __bf16* __restrict__ B,
    void* __restrict__ Cout, const float* __restrict__ R, int M, int N,
    int K) {
  const int tid  = threadIdx.x;
  const int lane = tid & 31;
  const int wave = tid >> 5;
  const int wrow = wave >> 1;  // 0..3
  const int wcol = wave & 1;   // 0..1
  const int bm = blockIdx.y * 128;
  const int bn = blockIdx.x * 128;
  const int m0 = lane & 15;
  const int kh = (lane >> 4) * 8;  // K half-lane split (ISA 7.12.2)

  __shared__ __bf16 As[128 * 40];
  __shared__ __bf16 Bs[128 * 40];

  // per-thread global-load slots: 512 chunks of 8 bf16 per tile, 2 per thread
  int ldrow[2], ldkc[2];
#pragma unroll
  for (int i = 0; i < 2; ++i) {
    int c = tid + i * 256;
    ldrow[i] = c >> 2;
    ldkc[i] = (c & 3) * 8;
  }

  const v8f vzero = {0.f, 0.f, 0.f, 0.f, 0.f, 0.f, 0.f, 0.f};
  v8f acc[2][4];
#pragma unroll
  for (int rt = 0; rt < 2; ++rt)
#pragma unroll
    for (int ct = 0; ct < 4; ++ct) acc[rt][ct] = vzero;

  // preload first K-tile into registers
  v8bf areg[2], breg[2];
#pragma unroll
  for (int i = 0; i < 2; ++i) {
    areg[i] = *(const v8bf*)&A[(size_t)(bm + ldrow[i]) * K + ldkc[i]];
    breg[i] = *(const v8bf*)&B[(size_t)(bn + ldrow[i]) * K + ldkc[i]];
  }

  for (int k0 = 0; k0 < K; k0 += 32) {
    __syncthreads();
#pragma unroll
    for (int i = 0; i < 2; ++i) {
      *(v8bf*)&As[ldrow[i] * 40 + ldkc[i]] = areg[i];
      *(v8bf*)&Bs[ldrow[i] * 40 + ldkc[i]] = breg[i];
    }
    __syncthreads();

    // prefetch next K-tile while WMMAs run
    if (k0 + 32 < K) {
#pragma unroll
      for (int i = 0; i < 2; ++i) {
        areg[i] = *(const v8bf*)&A[(size_t)(bm + ldrow[i]) * K + k0 + 32 + ldkc[i]];
        breg[i] = *(const v8bf*)&B[(size_t)(bn + ldrow[i]) * K + k0 + 32 + ldkc[i]];
      }
    }

    v16bf af[2], bfr[4];
#pragma unroll
    for (int rt = 0; rt < 2; ++rt) {
      const __bf16* p = &As[(wrow * 32 + rt * 16 + m0) * 40 + kh];
#pragma unroll
      for (int e = 0; e < 8; ++e) af[rt][e] = p[e];
#pragma unroll
      for (int e = 0; e < 8; ++e) af[rt][8 + e] = p[16 + e];
    }
#pragma unroll
    for (int ct = 0; ct < 4; ++ct) {
      const __bf16* p = &Bs[(wcol * 64 + ct * 16 + m0) * 40 + kh];
#pragma unroll
      for (int e = 0; e < 8; ++e) bfr[ct][e] = p[e];
#pragma unroll
      for (int e = 0; e < 8; ++e) bfr[ct][8 + e] = p[16 + e];
    }
#pragma unroll
    for (int rt = 0; rt < 2; ++rt)
#pragma unroll
      for (int ct = 0; ct < 4; ++ct)
        acc[rt][ct] = wmma_bf16(af[rt], bfr[ct], acc[rt][ct]);
  }

  // straight-line epilogue (mode resolved at compile time)
  const int half = lane >> 4;
#pragma unroll
  for (int rt = 0; rt < 2; ++rt)
#pragma unroll
    for (int ct = 0; ct < 4; ++ct)
#pragma unroll
      for (int r = 0; r < 8; ++r) {
        int gm = bm + wrow * 32 + rt * 16 + half * 8 + r;
        int gn = bn + wcol * 64 + ct * 16 + m0;
        size_t idx = (size_t)gm * N + gn;
        float v = acc[rt][ct][r];
        if constexpr (MODE == 0) {
          ((__bf16*)Cout)[idx] = (__bf16)v;
        } else if constexpr (MODE == 1) {
          ((float*)Cout)[idx] = v;
        } else {
          ((float*)Cout)[idx] = v + R[idx];
        }
      }
}

// ---------------------------------------------------------------------------
// Flash attention w/ ALiBi + causal mask.
// grid = (SEQ/64 query tiles, NH heads), block = 128 thr = 4 waves.
// Each wave owns 16 query rows; key tiles of 32, online softmax in-register.
// q: [SEQ, HS] bf16 (head h at cols h*HD..); kv: [SEQ, 2*HD] bf16 (k | v).
// ---------------------------------------------------------------------------
__global__ __launch_bounds__(128) void attn_kernel(
    const __bf16* __restrict__ q, const __bf16* __restrict__ kv,
    __bf16* __restrict__ o) {
  const int qt = blockIdx.x;
  const int h = blockIdx.y;
  const int tid = threadIdx.x;
  const int wave = tid >> 5;
  const int lane = tid & 31;
  const int qbase = qt * 64;
  const int mrow = qbase + wave * 16;
  const int m0 = lane & 15;
  const int kh = (lane >> 4) * 8;
  const int half = lane >> 4;
  const float scale = 0.08838834764831845f;            // 1/sqrt(128)
  const float slope = exp2f(-0.25f * (float)(h + 1));  // alibi slope

  __shared__ __bf16 Ks[32 * 128];
  __shared__ __bf16 Vs[32 * 128];
  __shared__ __bf16 Ps[4][16 * 32];

  // Q A-fragments (16 rows x 128 d, 4 k-steps), loaded once.
  v16bf qa[4];
  {
    const __bf16* qr = q + (size_t)(mrow + m0) * HS + h * HD;
#pragma unroll
    for (int kk = 0; kk < 4; ++kk) {
      const __bf16* p = qr + kk * 32 + kh;
#pragma unroll
      for (int e = 0; e < 8; ++e) qa[kk][e] = p[e];
#pragma unroll
      for (int e = 0; e < 8; ++e) qa[kk][8 + e] = p[16 + e];
    }
  }

  const v8f vzero = {0.f, 0.f, 0.f, 0.f, 0.f, 0.f, 0.f, 0.f};
  v8f O[8];
#pragma unroll
  for (int ct = 0; ct < 8; ++ct) O[ct] = vzero;
  float mprev[8], lsum[8];
#pragma unroll
  for (int r = 0; r < 8; ++r) {
    mprev[r] = -1e30f;
    lsum[r] = 0.f;
  }

  const int ktiles = qt * 2 + 2;  // covers keys 0..qbase+63
  for (int kt = 0; kt < ktiles; ++kt) {
    const int tbase = kt * 32;
    __syncthreads();
    for (int i = tid; i < 512; i += 128) {  // 32x128 each for K and V
      int row = i >> 4;
      int col = (i & 15) * 8;
      const __bf16* src = kv + (size_t)(tbase + row) * (2 * HD);
      *(v8bf*)&Ks[row * 128 + col] = *(const v8bf*)&src[col];
      *(v8bf*)&Vs[row * 128 + col] = *(const v8bf*)&src[HD + col];
    }
    __syncthreads();

    if (tbase <= mrow + 15) {  // wave-uniform: EXEC stays all-ones for WMMA
      // S = Q * K^T : two 16x16 accumulators (keys 0-15, 16-31 of tile)
      v8f s0 = vzero, s1 = vzero;
#pragma unroll
      for (int kk = 0; kk < 4; ++kk) {
        v16bf b0, b1;
        const __bf16* kp0 = &Ks[m0 * 128 + kk * 32 + kh];
        const __bf16* kp1 = &Ks[(m0 + 16) * 128 + kk * 32 + kh];
#pragma unroll
        for (int e = 0; e < 8; ++e) {
          b0[e] = kp0[e];
          b1[e] = kp1[e];
        }
#pragma unroll
        for (int e = 0; e < 8; ++e) {
          b0[8 + e] = kp0[16 + e];
          b1[8 + e] = kp1[16 + e];
        }
        s0 = wmma_bf16(qa[kk], b0, s0);
        s1 = wmma_bf16(qa[kk], b1, s1);
      }

      // ALiBi bias + causal mask + online softmax (C layout: lane->n, vgpr->m)
      float pr0[8], pr1[8];
#pragma unroll
      for (int r = 0; r < 8; ++r) {
        int mg = mrow + half * 8 + r;
        int t0 = tbase + m0;
        int t1 = tbase + 16 + m0;
        float v0 = (t0 <= mg) ? (s0[r] * scale + slope * (float)(t0 - mg))
                              : -1e30f;
        float v1 = (t1 <= mg) ? (s1[r] * scale + slope * (float)(t1 - mg))
                              : -1e30f;
        float mx = fmaxf(v0, v1);
        mx = fmaxf(mx, __shfl_xor(mx, 1));
        mx = fmaxf(mx, __shfl_xor(mx, 2));
        mx = fmaxf(mx, __shfl_xor(mx, 4));
        mx = fmaxf(mx, __shfl_xor(mx, 8));
        float mnew = fmaxf(mprev[r], mx);
        float alpha = __expf(mprev[r] - mnew);
        float p0 = __expf(v0 - mnew);
        float p1 = __expf(v1 - mnew);
        float rs = p0 + p1;
        rs += __shfl_xor(rs, 1);
        rs += __shfl_xor(rs, 2);
        rs += __shfl_xor(rs, 4);
        rs += __shfl_xor(rs, 8);
        lsum[r] = lsum[r] * alpha + rs;
        mprev[r] = mnew;
#pragma unroll
        for (int ct = 0; ct < 8; ++ct) O[ct][r] *= alpha;  // rescale O rows
        pr0[r] = p0;
        pr1[r] = p1;
      }

      // transpose P (C layout -> A layout) through per-wave LDS staging
#pragma unroll
      for (int r = 0; r < 8; ++r) {
        Ps[wave][(half * 8 + r) * 32 + m0] = (__bf16)pr0[r];
        Ps[wave][(half * 8 + r) * 32 + 16 + m0] = (__bf16)pr1[r];
      }
      v16bf pa;
      {
        const __bf16* pp = &Ps[wave][m0 * 32 + kh];
#pragma unroll
        for (int e = 0; e < 8; ++e) pa[e] = pp[e];
#pragma unroll
        for (int e = 0; e < 8; ++e) pa[8 + e] = pp[16 + e];
      }

      // O += P * V  (contraction over 32 keys, 8 d-tiles of 16)
#pragma unroll
      for (int ct = 0; ct < 8; ++ct) {
        v16bf vb;
        const __bf16* vp = &Vs[ct * 16 + m0];
#pragma unroll
        for (int e = 0; e < 8; ++e) vb[e] = vp[(kh + e) * 128];
#pragma unroll
        for (int e = 0; e < 8; ++e) vb[8 + e] = vp[(16 + kh + e) * 128];
        O[ct] = wmma_bf16(pa, vb, O[ct]);
      }
    }
  }

#pragma unroll
  for (int ct = 0; ct < 8; ++ct)
#pragma unroll
    for (int r = 0; r < 8; ++r) {
      int mg = mrow + half * 8 + r;
      float val = O[ct][r] / lsum[r];
      o[(size_t)mg * HS + h * HD + ct * 16 + m0] = (__bf16)val;
    }
}

// ---------------------------------------------------------------------------
// act = silu(g) * u  from gu fp32 [SEQ, 2*I] -> bf16 [SEQ, I]
// 2D grid: y = sequence row, x*256+tid = column (no integer div/mod)
// ---------------------------------------------------------------------------
__global__ __launch_bounds__(256) void silu_mul_kernel(
    const float* __restrict__ gu, __bf16* __restrict__ act) {
  const int row = blockIdx.y;
  const int col = blockIdx.x * 256 + threadIdx.x;
  if (col >= II) return;
  const float* gr = gu + (size_t)row * (2 * II);
  float g = gr[col];
  float u = gr[II + col];
  float sg = g / (1.f + __expf(-g));
  act[(size_t)row * II + col] = (__bf16)(sg * u);
}

// ---------------------------------------------------------------------------
extern "C" void kernel_launch(void* const* d_in, const int* in_sizes, int n_in,
                              void* d_out, int out_size, void* d_ws,
                              size_t ws_size, hipStream_t stream) {
  (void)in_sizes; (void)n_in; (void)out_size; (void)ws_size;
  const float* hidden = (const float*)d_in[0];
  const float* ln1w   = (const float*)d_in[1];
  const float* ln2w   = (const float*)d_in[2];
  const float* Wq     = (const float*)d_in[3];
  const float* Wkv    = (const float*)d_in[4];
  const float* Wo     = (const float*)d_in[5];
  const float* Wgu    = (const float*)d_in[6];
  const float* Wdown  = (const float*)d_in[7];

  size_t off = 0;
  auto alloc = [&](size_t bytes) -> void* {
    void* p = (char*)d_ws + off;
    off += (bytes + 255) & ~(size_t)255;
    return p;
  };

  __bf16* wq_b    = (__bf16*)alloc((size_t)HS * HS * 2);
  __bf16* wkv_b   = (__bf16*)alloc((size_t)(2 * HD) * HS * 2);
  __bf16* wo_b    = (__bf16*)alloc((size_t)HS * HS * 2);
  __bf16* wgu_b   = (__bf16*)alloc((size_t)(2 * II) * HS * 2);
  __bf16* wdown_b = (__bf16*)alloc((size_t)HS * II * 2);
  __bf16* xln     = (__bf16*)alloc((size_t)SEQ * HS * 2);
  __bf16* q_b     = (__bf16*)alloc((size_t)SEQ * HS * 2);
  __bf16* kv_b    = (__bf16*)alloc((size_t)SEQ * (2 * HD) * 2);
  __bf16* attn_b  = (__bf16*)alloc((size_t)SEQ * HS * 2);
  float*  h_f     = (float*)alloc((size_t)SEQ * HS * 4);
  __bf16* yln     = (__bf16*)alloc((size_t)SEQ * HS * 2);
  float*  gu_f    = (float*)alloc((size_t)SEQ * (2 * II) * 4);
  __bf16* act_b   = (__bf16*)alloc((size_t)SEQ * II * 2);

  // 1) weights -> bf16
  cast_bf16_kernel<<<2048, 256, 0, stream>>>(Wq, wq_b, (size_t)HS * HS);
  cast_bf16_kernel<<<512, 256, 0, stream>>>(Wkv, wkv_b, (size_t)(2 * HD) * HS);
  cast_bf16_kernel<<<2048, 256, 0, stream>>>(Wo, wo_b, (size_t)HS * HS);
  cast_bf16_kernel<<<4096, 256, 0, stream>>>(Wgu, wgu_b, (size_t)(2 * II) * HS);
  cast_bf16_kernel<<<4096, 256, 0, stream>>>(Wdown, wdown_b, (size_t)HS * II);

  // 2) LN1
  ln_bf16_kernel<<<SEQ, 256, 0, stream>>>(hidden, ln1w, xln, HS);

  // 3) q / kv projections (bf16 out)
  gemm_bf16_kernel<0><<<dim3(HS / 128, SEQ / 128), 256, 0, stream>>>(
      xln, wq_b, q_b, nullptr, SEQ, HS, HS);
  gemm_bf16_kernel<0><<<dim3((2 * HD) / 128, SEQ / 128), 256, 0, stream>>>(
      xln, wkv_b, kv_b, nullptr, SEQ, 2 * HD, HS);

  // 4) attention (flash, ALiBi, causal)
  attn_kernel<<<dim3(SEQ / 64, NH), 128, 0, stream>>>(q_b, kv_b, attn_b);

  // 5) output projection + residual -> h (fp32 out + residual)
  gemm_bf16_kernel<2><<<dim3(HS / 128, SEQ / 128), 256, 0, stream>>>(
      attn_b, wo_b, h_f, hidden, SEQ, HS, HS);

  // 6) LN2
  ln_bf16_kernel<<<SEQ, 256, 0, stream>>>(h_f, ln2w, yln, HS);

  // 7) gate/up projection (fp32 out)
  gemm_bf16_kernel<1><<<dim3((2 * II) / 128, SEQ / 128), 256, 0, stream>>>(
      yln, wgu_b, gu_f, nullptr, SEQ, 2 * II, HS);

  // 8) silu(g)*u
  silu_mul_kernel<<<dim3((II + 255) / 256, SEQ), 256, 0, stream>>>(gu_f, act_b);

  // 9) down projection + residual -> d_out (fp32 out + residual)
  gemm_bf16_kernel<2><<<dim3(HS / 128, SEQ / 128), 256, 0, stream>>>(
      act_b, wdown_b, (float*)d_out, h_f, SEQ, HS, II);
}
